// BilinearInteractionLayer_83167746719774
// MI455X (gfx1250) — matrix operand order repaired and checked by reading, not compiled, not run on hardware
//
#include <hip/hip_runtime.h>

// out[b,p,f] = (sum_e emb[b,row(p),e] * W[p,e,f]) * emb[b,col(p),f] + bias[p,f]
//
// Strategy (MI455X):
//  - Preprocess once per launch: split fp32 operands into bf16 hi+lo pairs
//    (x = hi + lo). Weight is simultaneously transposed to [p][f][e] so both
//    WMMA fragments become contiguous global b128 loads.
//  - GEMM kernel: pure register-resident tiles, no LDS/barriers. 3-term
//    split-bf16 accumulation (Ah*Bh + Ah*Bl + Al*Bh) in fp32 via
//    v_wmma_f32_16x16x32_bf16 -> ~2^-16 relative error (near-fp32).

#define NUM_FIELDS 16
#define EMBED      256
#define BATCHSZ    2048
#define NUM_PAIRS  120

#define BM    128
#define BN    128
#define KSTEP 32

typedef __bf16 bf16_t;
typedef bf16_t v16bf __attribute__((ext_vector_type(16)));
typedef float  v8f   __attribute__((ext_vector_type(8)));
typedef unsigned short u16;

union V16BF {
    v16bf v;
    uint4 q[2];
};

__device__ __forceinline__ u16 bf16_rn(float x) {
    unsigned u = __builtin_bit_cast(unsigned, x);
    u += 0x7FFFu + ((u >> 16) & 1u);     // round-to-nearest-even
    return (u16)(u >> 16);
}
__device__ __forceinline__ float bf16_to_f32(u16 h) {
    unsigned u = ((unsigned)h) << 16;
    return __builtin_bit_cast(float, u);
}

// ---------------- preprocess: emb (fp32) -> bf16 hi/lo, same layout ----------
__global__ __launch_bounds__(256)
void cvt_emb_kernel(const float* __restrict__ emb,
                    u16* __restrict__ hi, u16* __restrict__ lo) {
    const size_t idx = ((size_t)blockIdx.x * 256 + threadIdx.x) * 4;
    float4 x = *(const float4*)(emb + idx);
    u16 h0 = bf16_rn(x.x), h1 = bf16_rn(x.y), h2 = bf16_rn(x.z), h3 = bf16_rn(x.w);
    u16 g0 = bf16_rn(x.x - bf16_to_f32(h0));
    u16 g1 = bf16_rn(x.y - bf16_to_f32(h1));
    u16 g2 = bf16_rn(x.z - bf16_to_f32(h2));
    u16 g3 = bf16_rn(x.w - bf16_to_f32(h3));
    uint2 hp = make_uint2((unsigned)h0 | ((unsigned)h1 << 16),
                          (unsigned)h2 | ((unsigned)h3 << 16));
    uint2 lp = make_uint2((unsigned)g0 | ((unsigned)g1 << 16),
                          (unsigned)g2 | ((unsigned)g3 << 16));
    *(uint2*)(hi + idx) = hp;
    *(uint2*)(lo + idx) = lp;
}

// ------ preprocess: weight [p][e][f] fp32 -> bf16 hi/lo transposed [p][f][e] -
#define TLDS 72   // padded LDS row stride (ushorts); 144B rows keep 16B-aligned reads
__global__ __launch_bounds__(256)
void cvt_wt_kernel(const float* __restrict__ weight,
                   u16* __restrict__ wtHi, u16* __restrict__ wtLo) {
    __shared__ __align__(16) u16 tHi[64 * TLDS];
    __shared__ __align__(16) u16 tLo[64 * TLDS];

    const int p  = blockIdx.x >> 4;          // pair
    const int tl = blockIdx.x & 15;          // 4x4 tiles of 64x64
    const int e0 = (tl >> 2) * 64;
    const int f0 = (tl & 3) * 64;

    // read coalesced along f, store transposed [f][e] into LDS
    const int fc = threadIdx.x & 63;
    const int eg = threadIdx.x >> 6;         // 0..3
    const float* src = weight + ((size_t)p * EMBED + e0) * EMBED + f0;
#pragma unroll
    for (int r = 0; r < 16; ++r) {
        int e = eg * 16 + r;
        float x = src[(size_t)e * EMBED + fc];
        u16 h = bf16_rn(x);
        tHi[fc * TLDS + e] = h;
        tLo[fc * TLDS + e] = bf16_rn(x - bf16_to_f32(h));
    }
    __syncthreads();

    // write coalesced along e
    const int fr = threadIdx.x >> 2;         // 0..63
    const int ec = (threadIdx.x & 3) * 16;   // 0,16,32,48
    const size_t drow = ((size_t)p * EMBED + f0 + fr) * EMBED + e0 + ec;
    uint4 h4a = *(const uint4*)&tHi[fr * TLDS + ec];
    uint4 h4b = *(const uint4*)&tHi[fr * TLDS + ec + 8];
    uint4 l4a = *(const uint4*)&tLo[fr * TLDS + ec];
    uint4 l4b = *(const uint4*)&tLo[fr * TLDS + ec + 8];
    *(uint4*)(wtHi + drow)     = h4a;
    *(uint4*)(wtHi + drow + 8) = h4b;
    *(uint4*)(wtLo + drow)     = l4a;
    *(uint4*)(wtLo + drow + 8) = l4b;
}

// ---------------- main GEMM: register-only tiles, no LDS ---------------------
__global__ __launch_bounds__(256)
void bilinear_wmma_kernel(const u16* __restrict__ embHi, const u16* __restrict__ embLo,
                          const u16* __restrict__ wtHi,  const u16* __restrict__ wtLo,
                          const float* __restrict__ emb,
                          const float* __restrict__ bias,
                          float* __restrict__ out) {
    const int pair  = blockIdx.y;
    const int mtile = blockIdx.x >> 1;
    const int ntile = blockIdx.x & 1;

    // pair -> (rowf, colf) of triu_indices(16, k=1)
    int rem = pair, rowf = 0;
    while (rem >= (NUM_FIELDS - 1 - rowf)) { rem -= NUM_FIELDS - 1 - rowf; ++rowf; }
    const int colf = rowf + 1 + rem;

    const int lane = threadIdx.x & 31;
    const int wv   = threadIdx.x >> 5;
    const int wm   = wv & 3;          // 4 waves along M (32 rows)
    const int wn   = wv >> 2;         // 2 waves along N (64 cols)
    const int l16  = lane & 15;
    const int lh   = lane >> 4;
    const int c0   = lh * 8;

    // A base: rows m = l%16 of subtile; chunks [k0+c0, +8) and [k0+16+c0, +8)
    const size_t abase = ((size_t)((mtile * BM + wm * 32 + l16) * NUM_FIELDS + rowf)) * EMBED + c0;
    const u16* aH = embHi + abase;
    const u16* aL = embLo + abase;
    // B base: col n = l%16; chunks [k0+lh*8, +8) and [k0+16+lh*8, +8) in [f][e] layout
    const size_t bbase = ((size_t)(pair * EMBED + ntile * BN + wn * 64 + l16)) * EMBED + lh * 8;
    const u16* bH = wtHi + bbase;
    const u16* bL = wtLo + bbase;

    v8f acc[2][4];
#pragma unroll
    for (int i = 0; i < 2; ++i)
#pragma unroll
        for (int j = 0; j < 4; ++j)
            acc[i][j] = (v8f)(0.0f);

#pragma unroll
    for (int s = 0; s < EMBED / KSTEP; ++s) {
        const int k0 = s * KSTEP;

        V16BF ahi[2], alo[2];
#pragma unroll
        for (int i = 0; i < 2; ++i) {
            const int ao = i * (16 * NUM_FIELDS * EMBED);   // +16 batch rows
            ahi[i].q[0] = *(const uint4*)(aH + ao + k0);
            ahi[i].q[1] = *(const uint4*)(aH + ao + k0 + 16);
            alo[i].q[0] = *(const uint4*)(aL + ao + k0);
            alo[i].q[1] = *(const uint4*)(aL + ao + k0 + 16);
        }

#pragma unroll
        for (int j = 0; j < 4; ++j) {
            const int bo = j * (16 * EMBED);                // +16 f columns
            V16BF bhi, blo;
            bhi.q[0] = *(const uint4*)(bH + bo + k0);
            bhi.q[1] = *(const uint4*)(bH + bo + k0 + 16);
            blo.q[0] = *(const uint4*)(bL + bo + k0);
            blo.q[1] = *(const uint4*)(bL + bo + k0 + 16);
#pragma unroll
            for (int i = 0; i < 2; ++i) {
                acc[i][j] = __builtin_amdgcn_wmma_f32_16x16x32_bf16(
                    false, ahi[i].v, false, bhi.v, (short)0, acc[i][j], false, false);
                acc[i][j] = __builtin_amdgcn_wmma_f32_16x16x32_bf16(
                    false, alo[i].v, false, bhi.v, (short)0, acc[i][j], false, false);
                acc[i][j] = __builtin_amdgcn_wmma_f32_16x16x32_bf16(
                    false, ahi[i].v, false, blo.v, (short)0, acc[i][j], false, false);
            }
        }
    }

    // Epilogue: out = acc * q + bias  (C/D layout: lane<16 -> M=r; lane>=16 -> M=8+r)
#pragma unroll
    for (int i = 0; i < 2; ++i) {
#pragma unroll
        for (int j = 0; j < 4; ++j) {
            const int f  = ntile * BN + wn * 64 + j * 16 + l16;
            const float bv = bias[pair * EMBED + f];
            const int mb = mtile * BM + wm * 32 + i * 16 + lh * 8;
            v8f c = acc[i][j];
#pragma unroll
            for (int r = 0; r < 8; ++r) {
                const int b = mb + r;
                const float qv = emb[(size_t)(b * NUM_FIELDS + colf) * EMBED + f];
                out[((size_t)b * NUM_PAIRS + pair) * EMBED + f] = c[r] * qv + bv;
            }
        }
    }
}

extern "C" void kernel_launch(void* const* d_in, const int* in_sizes, int n_in,
                              void* d_out, int out_size, void* d_ws, size_t ws_size,
                              hipStream_t stream) {
    (void)in_sizes; (void)n_in; (void)out_size; (void)ws_size;
    const float* emb    = (const float*)d_in[0];
    const float* weight = (const float*)d_in[1];
    const float* bias   = (const float*)d_in[2];
    float*       out    = (float*)d_out;

    // workspace partition (bf16 elements)
    const size_t embN = (size_t)BATCHSZ * NUM_FIELDS * EMBED;   // 8,388,608
    const size_t wtN  = (size_t)NUM_PAIRS * EMBED * EMBED;      // 7,864,320
    u16* embHi = (u16*)d_ws;
    u16* embLo = embHi + embN;
    u16* wtHi  = embLo + embN;
    u16* wtLo  = wtHi + wtN;                                    // total ~65 MB

    cvt_emb_kernel<<<dim3((unsigned)(embN / 4 / 256)), dim3(256), 0, stream>>>(emb, embHi, embLo);
    cvt_wt_kernel<<<dim3(NUM_PAIRS * 16), dim3(256), 0, stream>>>(weight, wtHi, wtLo);

    dim3 grid((BATCHSZ / BM) * (EMBED / BN), NUM_PAIRS);  // 32 x 120
    bilinear_wmma_kernel<<<grid, dim3(256), 0, stream>>>(embHi, embLo, wtHi, wtLo,
                                                         emb, bias, out);
}
